// TransformerGNNEdge_51032801411391
// MI455X (gfx1250) — matrix operand order repaired and checked by reading, not compiled, not run on hardware
//
#include <hip/hip_runtime.h>
#include <hip/hip_bf16.h>

#define L  3072
#define D  768
#define H  12
#define HD 64
#define TD (3*D)
#define NT 14

typedef __attribute__((ext_vector_type(16))) __bf16 bf16x16;
typedef __attribute__((ext_vector_type(8)))  float  f32x8;
typedef __attribute__((ext_vector_type(4)))  int    v4i;

#if __has_builtin(__builtin_amdgcn_global_load_async_to_lds_b128)
#define HAVE_ASYNC_LDS 1
typedef __attribute__((address_space(1))) v4i gv4i;   // global int4
typedef __attribute__((address_space(3))) v4i lv4i;   // LDS int4
#else
#define HAVE_ASYNC_LDS 0
#endif

// Load 2x8 contiguous f32 and convert to a 16-element bf16 fragment.
__device__ __forceinline__ void cvt16(const float* __restrict__ g0,
                                      const float* __restrict__ g1,
                                      bf16x16& d) {
  float4 t0 = ((const float4*)g0)[0];
  float4 t1 = ((const float4*)g0)[1];
  float4 t2 = ((const float4*)g1)[0];
  float4 t3 = ((const float4*)g1)[1];
  d[0]=(__bf16)t0.x;  d[1]=(__bf16)t0.y;  d[2]=(__bf16)t0.z;  d[3]=(__bf16)t0.w;
  d[4]=(__bf16)t1.x;  d[5]=(__bf16)t1.y;  d[6]=(__bf16)t1.z;  d[7]=(__bf16)t1.w;
  d[8]=(__bf16)t2.x;  d[9]=(__bf16)t2.y;  d[10]=(__bf16)t2.z; d[11]=(__bf16)t2.w;
  d[12]=(__bf16)t3.x; d[13]=(__bf16)t3.y; d[14]=(__bf16)t3.z; d[15]=(__bf16)t3.w;
}

// Load 2x8 contiguous bf16 into a fragment.
__device__ __forceinline__ void ld16bf(const __bf16* __restrict__ p0,
                                       const __bf16* __restrict__ p1,
                                       bf16x16& d) {
  #pragma unroll
  for (int e = 0; e < 8; ++e) { d[e] = p0[e]; d[e+8] = p1[e]; }
}

// ---------------- Kernel A: qkv = x @ Win^T + bin, split into Q/K/V bf16 ----
// Block tile 128(M) x 128(N): 8 waves in 4(M) x 2(N) grid, 32x64 per wave.
__global__ __launch_bounds__(256)
void qkv_gemm_kernel(const float* __restrict__ x, const float* __restrict__ W,
                     const float* __restrict__ bias,
                     __bf16* __restrict__ Qb, __bf16* __restrict__ Kb,
                     __bf16* __restrict__ Vb)
{
  const int wave = threadIdx.x >> 5, lane = threadIdx.x & 31;
  const int bm = blockIdx.x / (TD/128);
  const int bn = blockIdx.x % (TD/128);
  const int m0 = bm*128 + (wave >> 1)*32;
  const int n0 = bn*128 + (wave & 1)*64;
  const int mn = lane & 15, half = lane >> 4;

  const float* ar0 = x + (size_t)(m0 + mn) * D;
  const float* ar1 = x + (size_t)(m0 + 16 + mn) * D;
  const float* br0 = W + (size_t)(n0 + mn) * D;        // +nt*16*D per N-subtile

  f32x8 c[2][4];
  #pragma unroll
  for (int mi = 0; mi < 2; ++mi)
    #pragma unroll
    for (int nt = 0; nt < 4; ++nt)
      c[mi][nt] = (f32x8){0.f,0.f,0.f,0.f,0.f,0.f,0.f,0.f};

  for (int kb = 0; kb < D; kb += 32) {
    bf16x16 a0, a1;
    cvt16(ar0 + kb + half*8, ar0 + kb + 16 + half*8, a0);
    cvt16(ar1 + kb + half*8, ar1 + kb + 16 + half*8, a1);
    #pragma unroll
    for (int nt = 0; nt < 4; ++nt) {
      bf16x16 b;
      const float* bp = br0 + (size_t)nt*16*D + kb + half*8;
      cvt16(bp, bp + 16, b);
      c[0][nt] = __builtin_amdgcn_wmma_f32_16x16x32_bf16(false, a0, false, b, (short)0, c[0][nt], false, false);
      c[1][nt] = __builtin_amdgcn_wmma_f32_16x16x32_bf16(false, a1, false, b, (short)0, c[1][nt], false, false);
    }
  }

  #pragma unroll
  for (int nt = 0; nt < 4; ++nt) {
    const int j = n0 + nt*16 + mn;
    const float bj = bias[j];
    __bf16* dst; int jj; float scale = 1.0f;
    if (j < D)        { dst = Qb; jj = j;       scale = 0.125f; }   // 1/sqrt(64) folded into Q
    else if (j < 2*D) { dst = Kb; jj = j - D;   }
    else              { dst = Vb; jj = j - 2*D; }
    const int hh = jj >> 6, hd = jj & 63;
    __bf16* op = dst + ((size_t)hh * L) * HD + hd;
    #pragma unroll
    for (int mi = 0; mi < 2; ++mi)
      #pragma unroll
      for (int v = 0; v < 8; ++v) {
        int row = m0 + mi*16 + v + 8*half;
        op[(size_t)row * HD] = (__bf16)((c[mi][nt][v] + bj) * scale);
      }
  }
}

// ---------------- Kernel B: fused flash attention with edge bias ------------
__global__ __launch_bounds__(256)
void attn_kernel(const __bf16* __restrict__ Qb, const __bf16* __restrict__ Kb,
                 const __bf16* __restrict__ Vb,
                 const float* __restrict__ mask, const int* __restrict__ etype,
                 const float* __restrict__ type_emb,
                 __bf16* __restrict__ AO)
{
  __shared__ __align__(16) __bf16 Kch[2][32*HD];   // double-buffered 32-key chunks
  __shared__ __align__(16) __bf16 Vch[2][32*HD];
  __shared__ __align__(16) __bf16 Pst[8*16*32];    // per-wave P transpose staging
  __shared__ float temb[NT*H];

  const int wave = threadIdx.x >> 5, lane = threadIdx.x & 31;
  const int h  = blockIdx.x / (L/128);
  const int q0 = (blockIdx.x % (L/128)) * 128 + wave * 16;
  const int n  = lane & 15, half = lane >> 4;

  if (threadIdx.x < NT*H) temb[threadIdx.x] = type_emb[threadIdx.x];

  const __bf16* Kg = Kb + (size_t)h * L * HD;
  const __bf16* Vg = Vb + (size_t)h * L * HD;

  // Q fragment (A operand), both 32-wide halves of head_dim (pre-scaled).
  const __bf16* qrow = Qb + ((size_t)h * L + (q0 + n)) * HD;
  bf16x16 aq0, aq1;
  #pragma unroll
  for (int e = 0; e < 8; ++e) {
    aq0[e]   = qrow[half*16 + e];       aq0[e+8] = qrow[half*16 + 8 + e];
    aq1[e]   = qrow[32 + half*16 + e];  aq1[e+8] = qrow[40 + half*16 + e];
  }

  f32x8 o[4];
  #pragma unroll
  for (int nt = 0; nt < 4; ++nt) o[nt] = (f32x8){0.f,0.f,0.f,0.f,0.f,0.f,0.f,0.f};
  float rowmax[8], rowsum[8];
  #pragma unroll
  for (int v = 0; v < 8; ++v) { rowmax[v] = -__builtin_inff(); rowsum[v] = 0.f; }

  // ---- K/V chunk staging: async copy to LDS (ASYNCcnt path) when available.
  const int tid = threadIdx.x;
#if HAVE_ASYNC_LDS
  #define STAGE_KV(buf, j)                                                          \
    do {                                                                            \
      __builtin_amdgcn_global_load_async_to_lds_b128(                               \
          (gv4i*)(Kg + (size_t)(j)*HD + tid*8),                                     \
          (lv4i*)(&Kch[buf][tid*8]), 0, 0);                                         \
      __builtin_amdgcn_global_load_async_to_lds_b128(                               \
          (gv4i*)(Vg + (size_t)(j)*HD + tid*8),                                     \
          (lv4i*)(&Vch[buf][tid*8]), 0, 0);                                         \
    } while (0)
  #define WAIT_ASYNC() __builtin_amdgcn_s_wait_asynccnt(0)
#else
  #define STAGE_KV(buf, j)                                                          \
    do {                                                                            \
      ((uint4*)Kch[buf])[tid] = ((const uint4*)(Kg + (size_t)(j)*HD))[tid];         \
      ((uint4*)Vch[buf])[tid] = ((const uint4*)(Vg + (size_t)(j)*HD))[tid];         \
    } while (0)
  #define WAIT_ASYNC() do {} while (0)
#endif

  STAGE_KV(0, 0);   // prologue

  for (int j0 = 0; j0 < L; j0 += 32) {
    const int cur = (j0 >> 5) & 1;
    WAIT_ASYNC();        // this wave's copy into buffer `cur` complete
    __syncthreads();     // all waves' copies visible; prev iteration's reads done
    if (j0 + 32 < L) {
      STAGE_KV(cur ^ 1, j0 + 32);   // overlap next chunk with compute
      // prefetch next mask/edge_types block (dominant HBM stream)
      const int prow = q0 + n, pcol = j0 + 32 + half*16;
      __builtin_prefetch(&mask[(size_t)prow*L + pcol], 0, 0);
      __builtin_prefetch(&etype[(size_t)prow*L + pcol], 0, 0);
    }
    const __bf16* Kc = Kch[cur];
    const __bf16* Vc = Vch[cur];

    // scores: two 16x16 key tiles, contraction over hd=64 (2 WMMA steps each)
    f32x8 s0 = {0.f,0.f,0.f,0.f,0.f,0.f,0.f,0.f};
    f32x8 s1 = {0.f,0.f,0.f,0.f,0.f,0.f,0.f,0.f};
    bf16x16 b00, b01, b10, b11;
    const __bf16* kr0 = Kc + (size_t)n * HD;         // key row j0+n
    const __bf16* kr1 = Kc + (size_t)(16+n) * HD;    // key row j0+16+n
    #pragma unroll
    for (int e = 0; e < 16; ++e) {
      b00[e] = kr0[half*16 + e];       b01[e] = kr0[32 + half*16 + e];
      b10[e] = kr1[half*16 + e];       b11[e] = kr1[32 + half*16 + e];
    }
    s0 = __builtin_amdgcn_wmma_f32_16x16x32_bf16(false, aq0, false, b00, (short)0, s0, false, false);
    s0 = __builtin_amdgcn_wmma_f32_16x16x32_bf16(false, aq1, false, b01, (short)0, s0, false, false);
    s1 = __builtin_amdgcn_wmma_f32_16x16x32_bf16(false, aq0, false, b10, (short)0, s1, false, false);
    s1 = __builtin_amdgcn_wmma_f32_16x16x32_bf16(false, aq1, false, b11, (short)0, s1, false, false);

    // edge-type bias + additive mask, element-wise in C layout
    #pragma unroll
    for (int v = 0; v < 8; ++v) {
      int q   = q0 + v + 8*half;
      int k0i = j0 + n, k1i = j0 + 16 + n;
      s0[v] += mask[(size_t)q*L + k0i] + temb[etype[(size_t)q*L + k0i]*H + h];
      s1[v] += mask[(size_t)q*L + k1i] + temb[etype[(size_t)q*L + k1i]*H + h];
    }

    // online softmax over the 32 keys; row m lives across a 16-lane half
    __bf16* pw = Pst + wave*(16*32);
    #pragma unroll
    for (int v = 0; v < 8; ++v) {
      float mx = fmaxf(s0[v], s1[v]);
      #pragma unroll
      for (int off = 1; off < 16; off <<= 1) mx = fmaxf(mx, __shfl_xor(mx, off, 32));
      float mnew  = fmaxf(rowmax[v], mx);
      float alpha = __expf(rowmax[v] - mnew);
      float p0 = __expf(s0[v] - mnew);
      float p1 = __expf(s1[v] - mnew);
      float ps = p0 + p1;
      #pragma unroll
      for (int off = 1; off < 16; off <<= 1) ps += __shfl_xor(ps, off, 32);
      rowsum[v] = rowsum[v]*alpha + ps;
      rowmax[v] = mnew;
      o[0][v] *= alpha; o[1][v] *= alpha; o[2][v] *= alpha; o[3][v] *= alpha;
      int mrow = v + 8*half;
      pw[mrow*32 + n]      = (__bf16)p0;      // P[m][k] row-major in LDS
      pw[mrow*32 + 16 + n] = (__bf16)p1;
    }
    __syncthreads();   // P visible for A-layout reads

    // O += P(16x32) @ V(32x64): A from LDS transpose, B gathered from Vch
    bf16x16 ap;
    const __bf16* pr = Pst + wave*(16*32) + n*32;
    #pragma unroll
  for (int e = 0; e < 8; ++e) { ap[e] = pr[half*16 + e]; ap[e+8] = pr[half*16 + 8 + e]; }
    #pragma unroll
    for (int nt = 0; nt < 4; ++nt) {
      bf16x16 bv;
      #pragma unroll
      for (int e = 0; e < 16; ++e) bv[e] = Vc[(size_t)(half*16 + e)*HD + nt*16 + n];
      o[nt] = __builtin_amdgcn_wmma_f32_16x16x32_bf16(false, ap, false, bv,
                                                      (short)0, o[nt], false, false);
    }
  }

  // normalize and merge heads into (L, D) bf16
  #pragma unroll
  for (int v = 0; v < 8; ++v) {
    float inv = 1.0f / rowsum[v];
    int q = q0 + v + 8*half;
    #pragma unroll
    for (int nt = 0; nt < 4; ++nt)
      AO[(size_t)q*D + h*HD + nt*16 + n] = (__bf16)(o[nt][v] * inv);
  }
}

// ---------------- Kernel C: out = AO @ Wout^T + bout (fp32 result) ----------
// Same 128x128 block tile, 32x64 per wave.
__global__ __launch_bounds__(256)
void out_gemm_kernel(const __bf16* __restrict__ AO, const float* __restrict__ W,
                     const float* __restrict__ bias, float* __restrict__ out)
{
  const int wave = threadIdx.x >> 5, lane = threadIdx.x & 31;
  const int bm = blockIdx.x / (D/128);
  const int bn = blockIdx.x % (D/128);
  const int m0 = bm*128 + (wave >> 1)*32;
  const int n0 = bn*128 + (wave & 1)*64;
  const int mn = lane & 15, half = lane >> 4;

  const __bf16* ar0 = AO + (size_t)(m0 + mn) * D;
  const __bf16* ar1 = AO + (size_t)(m0 + 16 + mn) * D;
  const float*  br0 = W  + (size_t)(n0 + mn) * D;

  f32x8 c[2][4];
  #pragma unroll
  for (int mi = 0; mi < 2; ++mi)
    #pragma unroll
    for (int nt = 0; nt < 4; ++nt)
      c[mi][nt] = (f32x8){0.f,0.f,0.f,0.f,0.f,0.f,0.f,0.f};

  for (int kb = 0; kb < D; kb += 32) {
    bf16x16 a0, a1;
    ld16bf(ar0 + kb + half*8, ar0 + kb + 16 + half*8, a0);
    ld16bf(ar1 + kb + half*8, ar1 + kb + 16 + half*8, a1);
    #pragma unroll
    for (int nt = 0; nt < 4; ++nt) {
      bf16x16 b;
      const float* bp = br0 + (size_t)nt*16*D + kb + half*8;
      cvt16(bp, bp + 16, b);
      c[0][nt] = __builtin_amdgcn_wmma_f32_16x16x32_bf16(false, a0, false, b, (short)0, c[0][nt], false, false);
      c[1][nt] = __builtin_amdgcn_wmma_f32_16x16x32_bf16(false, a1, false, b, (short)0, c[1][nt], false, false);
    }
  }

  #pragma unroll
  for (int nt = 0; nt < 4; ++nt) {
    const float bj = bias[n0 + nt*16 + mn];
    #pragma unroll
    for (int mi = 0; mi < 2; ++mi)
      #pragma unroll
      for (int v = 0; v < 8; ++v)
        out[(size_t)(m0 + mi*16 + v + 8*half)*D + n0 + nt*16 + mn] = c[mi][nt][v] + bj;
  }
}

extern "C" void kernel_launch(void* const* d_in, const int* in_sizes, int n_in,
                              void* d_out, int out_size, void* d_ws, size_t ws_size,
                              hipStream_t stream) {
  const float* x    = (const float*)d_in[0];
  const float* mask = (const float*)d_in[1];
  const int*   et   = (const int*)d_in[2];
  const float* Win  = (const float*)d_in[3];
  const float* bin  = (const float*)d_in[4];
  const float* Wout = (const float*)d_in[5];
  const float* bout = (const float*)d_in[6];
  const float* temb = (const float*)d_in[7];
  float* out = (float*)d_out;

  __bf16* Qb = (__bf16*)d_ws;                    // (H, L, HD) bf16, pre-scaled
  __bf16* Kb = Qb + (size_t)H*L*HD;
  __bf16* Vb = Kb + (size_t)H*L*HD;
  __bf16* AO = Vb + (size_t)H*L*HD;              // (L, D) bf16 attention output

  qkv_gemm_kernel<<<(L/128)*(TD/128), 256, 0, stream>>>(x, Win, bin, Qb, Kb, Vb);
  attn_kernel<<<H*(L/128), 256, 0, stream>>>(Qb, Kb, Vb, mask, et, temb, AO);
  out_gemm_kernel<<<(L/128)*(D/128), 256, 0, stream>>>(AO, Wout, bout, out);
}
